// LSTMModel_82798379532634
// MI455X (gfx1250) — matrix operand order repaired and checked by reading, not compiled, not run on hardware
//
#include <hip/hip_runtime.h>
#include <hip/hip_bf16.h>

typedef __attribute__((ext_vector_type(16))) _Float16 v16h;
typedef __attribute__((ext_vector_type(4)))  _Float16 h4;
typedef __attribute__((ext_vector_type(8)))  float    v8f;
typedef __attribute__((ext_vector_type(4)))  float    f32x4;
typedef __attribute__((ext_vector_type(4)))  unsigned u32x4;
typedef __attribute__((ext_vector_type(4)))  int      i32x4;
typedef __attribute__((ext_vector_type(8)))  int      i32x8;

__device__ __forceinline__ v8f wmma16(v16h a, v16h b, v8f c) {
    // (neg_a, A, neg_b, B, c_mod, C, reuse_a, reuse_b)
    return __builtin_amdgcn_wmma_f32_16x16x32_f16(false, a, false, b, (short)0, c, false, false);
}
// Fast sigmoid/tanh: v_exp_f32 + v_rcp_f32 (TRANS, co-execute with XDL WMMA).
__device__ __forceinline__ float frcp(float x)  { return __builtin_amdgcn_rcpf(x); }
__device__ __forceinline__ float fsig(float x)  { return frcp(1.0f + __expf(-x)); }
__device__ __forceinline__ float ftanh(float x) { return 1.0f - 2.0f * frcp(__expf(2.0f * x) + 1.0f); }
__device__ __forceinline__ v8f splat8(float x) {
    v8f v;
#pragma unroll
    for (int i = 0; i < 8; ++i) v[i] = x;
    return v;
}
#define RFL(x) __builtin_amdgcn_readfirstlane((int)(x))

// ---- Tensor Data Mover: 2D f16 tile (rows x width), packed into LDS -----------
// D# group0: count=1 | lds_addr | global_addr[56:0] | type=2
// D# group1: data_size=2B | tensor_dim0/1 | tile_dim0/1 | tensor_dim0_stride
__device__ __forceinline__ void tdm_load_2d_f16(unsigned lds_byte_off,
                                                unsigned long long gaddr,
                                                unsigned width_elems, unsigned rows,
                                                unsigned long long row_stride_elems) {
    u32x4 g0 = {0u, 0u, 0u, 0u};
    g0[0] = (unsigned)RFL(1u);
    g0[1] = (unsigned)RFL(lds_byte_off);
    g0[2] = (unsigned)RFL((unsigned)gaddr);
    g0[3] = (unsigned)RFL((((unsigned)(gaddr >> 32)) & 0x01FFFFFFu) | 0x80000000u);
    i32x8 g1 = {0, 0, 0, 0, 0, 0, 0, 0};
    g1[0] = RFL(1u << 16);                                            // data_size = 2 bytes
    g1[1] = RFL((width_elems & 0xFFFFu) << 16);                       // tensor_dim0 lo16
    g1[2] = RFL((width_elems >> 16) | ((rows & 0xFFFFu) << 16));      // td0 hi | td1 lo
    g1[3] = RFL((rows >> 16) | ((width_elems & 0xFFFFu) << 16));      // td1 hi | tile_dim0
    g1[4] = RFL(rows & 0xFFFFu);                                      // tile_dim1 (tile_dim2=0)
    g1[5] = RFL((unsigned)row_stride_elems);                          // dim0_stride lo32
    g1[6] = RFL(((unsigned)(row_stride_elems >> 32)) & 0xFFFFu);      // dim0_stride hi16
    i32x4 z4 = {0, 0, 0, 0};
#if defined(__clang_major__) && (__clang_major__ >= 23)
    i32x8 z8 = {0, 0, 0, 0, 0, 0, 0, 0};
    __builtin_amdgcn_tensor_load_to_lds(g0, g1, z4, z4, z8, 0);
#else
    __builtin_amdgcn_tensor_load_to_lds(g0, g1, z4, z4, 0);
#endif
}

// Build a 16x32 f16 A-fragment from a row-major f16 LDS tile (combined layout).
template <int STRIDE>
__device__ __forceinline__ v16h load_a_frag(const _Float16* base, int m, int kbase, int lane) {
    v16h a;
    const int hi = (lane >= 16) ? 8 : 0;
#pragma unroll
    for (int e = 0; e < 16; e += 2) {
        const int r  = e >> 1;
        const int kk = kbase + ((r & 4) ? 16 : 0) + hi + (r & 3) * 2;
        a[e]     = base[m * STRIDE + kk];
        a[e + 1] = base[m * STRIDE + kk + 1];
    }
    return a;
}

// Split-source A-fragment: columns < F come from the x-chunk region, the rest
// from the [h | zero-pad] staging region. F%16==0 => each 8-half run stays in
// one region, so runs still merge into ds_load_b128.
template <int F, int HS, int TcF>
__device__ __forceinline__ v16h make_a_frag(const _Float16* xrow, const _Float16* hst,
                                            int m, int kbase, int lane) {
    v16h a;
    const int hi = (lane >= 16) ? 8 : 0;
#pragma unroll
    for (int run = 0; run < 2; ++run) {
        const int blk = kbase + run * 16;          // 16-aligned block
        const int cb  = blk + hi;
        const _Float16* p = (blk < F) ? (xrow + m * TcF + cb)
                                      : (hst + m * HS + (cb - F));
#pragma unroll
        for (int q = 0; q < 8; ++q) a[run * 8 + q] = p[q];
    }
    return a;
}

// Shared pieces of the LSTM cell --------------------------------------------------

// ===== Layer 1: f32 input, register-pipelined float4 staging =====
template <int H, int F, int WAVES>
__global__ __launch_bounds__(32 * WAVES)
void lstm_layer_f32_kernel(const float* __restrict__ xf,
                           const float* __restrict__ Wih, const float* __restrict__ Whh,
                           const float* __restrict__ bih, const float* __restrict__ bhh,
                           _Float16* __restrict__ seq_out, int T) {
    constexpr int NT = 4 * H / 16;
    constexpr int NH = H / 16;
    constexpr int NC  = 16 * F / 4;        // float4 chunks per tile
    constexpr int NIT = (NC + 31) / 32;    // per-lane chunk slots
    constexpr int CPR = F / 4;             // chunks per row
    static_assert(F + H <= 64, "");

    __shared__ _Float16 axh_all[WAVES][16 * 64];

    const int lane = threadIdx.x & 31;
    const int wave = threadIdx.x >> 5;
    const int b0   = (blockIdx.x * WAVES + wave) * 16;
    const int nloc = lane & 15;
    const int m8   = (lane >> 4) << 3;
    _Float16* axh  = axh_all[wave];

    for (int i = lane; i < 16 * 64; i += 32) axh[i] = (_Float16)0.0f;

    // Weight B-fragments: Wcat[k,n] = Wih[n,k] | Whh[n,k-F] | 0
    v16h Bf[2][NT];
#pragma unroll
    for (int s = 0; s < 2; ++s) {
        const int kh = s * 32 + ((lane >= 16) ? 8 : 0);
#pragma unroll
        for (int nt = 0; nt < NT; ++nt)
#pragma unroll
            for (int e = 0; e < 16; ++e) {
                const int n = nt * 16 + (lane & 15);
                const int k = kh + ((e & 8) ? 16 : 0) + (e & 7);
                float w = 0.0f;
                if (k < F)          w = Wih[n * F + k];
                else if (k < F + H) w = Whh[n * H + (k - F)];
                Bf[s][nt][e] = (_Float16)w;
            }
    }
    float bias[NT];
#pragma unroll
    for (int nt = 0; nt < NT; ++nt) bias[nt] = bih[nt * 16 + nloc] + bhh[nt * 16 + nloc];

    size_t gx[NIT]; int lx[NIT];
#pragma unroll
    for (int i = 0; i < NIT; ++i) {
        const int c = lane + i * 32;
        const int m = (c < NC) ? c / CPR : 0;
        const int q = (c < NC) ? c - m * CPR : 0;
        gx[i] = (size_t)(b0 + m) * T * F + 4 * q;
        lx[i] = m * 64 + 4 * q;
    }
    size_t hrow[8];
#pragma unroll
    for (int r = 0; r < 8; ++r) hrow[r] = (size_t)(b0 + r + m8) * T;

    v8f cst[NH], hreg[NH];
#pragma unroll
    for (int j = 0; j < NH; ++j) { cst[j] = splat8(0.0f); hreg[j] = splat8(0.0f); }

    // Prologue: stage x_0 (float4 loads -> f16 LDS stores).
    f32x4 xr[NIT];
#pragma unroll
    for (int i = 0; i < NIT; ++i)
        if (NC % 32 == 0 || lane + i * 32 < NC) xr[i] = *(const f32x4*)(xf + gx[i]);
#pragma unroll
    for (int i = 0; i < NIT; ++i)
        if (NC % 32 == 0 || lane + i * 32 < NC) {
            h4 hv = {(_Float16)xr[i][0], (_Float16)xr[i][1], (_Float16)xr[i][2], (_Float16)xr[i][3]};
            *(h4*)(axh + lx[i]) = hv;
        }

    for (int t = 0; t < T; ++t) {
        const v16h A0 = load_a_frag<64>(axh, nloc, 0, lane);
        const v16h A1 = load_a_frag<64>(axh, nloc, 32, lane);

        if (t + 1 < T) {
            const size_t toff = (size_t)(t + 1) * F;
#pragma unroll
            for (int i = 0; i < NIT; ++i)
                if (NC % 32 == 0 || lane + i * 32 < NC) xr[i] = *(const f32x4*)(xf + gx[i] + toff);
        }

        v8f acc[NT];
#pragma unroll
        for (int nt = 0; nt < NT; ++nt) {
            acc[nt] = splat8(bias[nt]);
            acc[nt] = wmma16(A0, Bf[0][nt], acc[nt]);
            acc[nt] = wmma16(A1, Bf[1][nt], acc[nt]);
        }
#pragma unroll
        for (int j = 0; j < NH; ++j) {
            const v8f ig = acc[0 * NH + j], fg = acc[1 * NH + j];
            const v8f gg = acc[2 * NH + j], og = acc[3 * NH + j];
#pragma unroll
            for (int r = 0; r < 8; ++r) {
                const float cv = fsig(fg[r]) * cst[j][r] + fsig(ig[r]) * ftanh(gg[r]);
                cst[j][r]  = cv;
                hreg[j][r] = fsig(og[r]) * ftanh(cv);
            }
        }
#pragma unroll
        for (int j = 0; j < NH; ++j)
#pragma unroll
            for (int r = 0; r < 8; ++r) {
                const _Float16 hv = (_Float16)hreg[j][r];
                axh[(r + m8) * 64 + F + j * 16 + nloc] = hv;
                seq_out[(hrow[r] + t) * H + j * 16 + nloc] = hv;
            }
        if (t + 1 < T) {
#pragma unroll
            for (int i = 0; i < NIT; ++i)
                if (NC % 32 == 0 || lane + i * 32 < NC) {
                    h4 hv = {(_Float16)xr[i][0], (_Float16)xr[i][1], (_Float16)xr[i][2], (_Float16)xr[i][3]};
                    *(h4*)(axh + lx[i]) = hv;
                }
        }
    }
}

// ===== Layers 2/3: f16 input streamed by the Tensor Data Mover =====
// One TDM per Tc timesteps (chunk = 16 rows x Tc*F halves, row stride T*F),
// double-buffered; h lives in a separate [h|zero] staging region.
template <int H, int F, int Tc, bool WRITE_SEQ, int WAVES>
__global__ __launch_bounds__(32 * WAVES)
void lstm_layer_tdm_kernel(const _Float16* __restrict__ xin,
                           const float* __restrict__ Wih, const float* __restrict__ Whh,
                           const float* __restrict__ bih, const float* __restrict__ bhh,
                           _Float16* __restrict__ seq_out, float* __restrict__ hfinal, int T) {
    constexpr int NT  = 4 * H / 16;
    constexpr int NH  = H / 16;
    constexpr int HS  = 64 - F;          // staging columns for [h | zero-pad]
    constexpr int TcF = Tc * F;
    static_assert(F % 16 == 0 && F + H <= 64, "");

    __shared__ _Float16 xch_all[WAVES][2][16 * TcF];   // declared first: LDS offset 0
    __shared__ _Float16 hst_all[WAVES][16 * HS];

    const int lane  = threadIdx.x & 31;
    const int waveu = __builtin_amdgcn_readfirstlane((int)(threadIdx.x >> 5));
    const int b0    = (blockIdx.x * WAVES + waveu) * 16;
    const int nloc  = lane & 15;
    const int m8    = (lane >> 4) << 3;
    _Float16* hst   = hst_all[waveu];

    for (int i = lane; i < 16 * HS; i += 32) hst[i] = (_Float16)0.0f;

    v16h Bf[2][NT];
#pragma unroll
    for (int s = 0; s < 2; ++s) {
        const int kh = s * 32 + ((lane >= 16) ? 8 : 0);
#pragma unroll
        for (int nt = 0; nt < NT; ++nt)
#pragma unroll
            for (int e = 0; e < 16; ++e) {
                const int n = nt * 16 + (lane & 15);
                const int k = kh + ((e & 8) ? 16 : 0) + (e & 7);
                float w = 0.0f;
                if (k < F)          w = Wih[n * F + k];
                else if (k < F + H) w = Whh[n * H + (k - F)];
                Bf[s][nt][e] = (_Float16)w;
            }
    }
    float bias[NT];
#pragma unroll
    for (int nt = 0; nt < NT; ++nt) bias[nt] = bih[nt * 16 + nloc] + bhh[nt * 16 + nloc];

    size_t hrow[8];
#pragma unroll
    for (int r = 0; r < 8; ++r) hrow[r] = (size_t)(b0 + r + m8) * T;

    v8f cst[NH], hreg[NH];
#pragma unroll
    for (int j = 0; j < NH; ++j) { cst[j] = splat8(0.0f); hreg[j] = splat8(0.0f); }

    const unsigned long long xbase = (unsigned long long)(size_t)xin
                                   + 2ull * (size_t)b0 * T * F;
    const unsigned xlds0   = (unsigned)(waveu * 2) * (unsigned)(16 * TcF) * 2u;
    const int      nchunk  = T / Tc;

    // Prologue: DMA chunk 0 into buffer 0.
    tdm_load_2d_f16(xlds0, xbase, TcF, 16, (unsigned long long)T * F);

    for (int t = 0; t < T; ++t) {
        const int c    = t / Tc;
        const int tmod = t - c * Tc;
        if (tmod == 0) {
            __builtin_amdgcn_s_wait_tensorcnt(0);          // chunk c has landed
            if (c + 1 < nchunk)                            // kick chunk c+1 into the other buffer
                tdm_load_2d_f16(xlds0 + (unsigned)(((c + 1) & 1) * (16 * TcF * 2)),
                                xbase + 2ull * (size_t)(c + 1) * TcF,
                                TcF, 16, (unsigned long long)T * F);
        }
        const _Float16* xrow = xch_all[waveu][c & 1] + tmod * F;
        const v16h A0 = make_a_frag<F, HS, TcF>(xrow, hst, nloc, 0, lane);
        const v16h A1 = make_a_frag<F, HS, TcF>(xrow, hst, nloc, 32, lane);

        v8f acc[NT];
#pragma unroll
        for (int nt = 0; nt < NT; ++nt) {
            acc[nt] = splat8(bias[nt]);
            acc[nt] = wmma16(A0, Bf[0][nt], acc[nt]);
            acc[nt] = wmma16(A1, Bf[1][nt], acc[nt]);
        }
#pragma unroll
        for (int j = 0; j < NH; ++j) {
            const v8f ig = acc[0 * NH + j], fg = acc[1 * NH + j];
            const v8f gg = acc[2 * NH + j], og = acc[3 * NH + j];
#pragma unroll
            for (int r = 0; r < 8; ++r) {
                const float cv = fsig(fg[r]) * cst[j][r] + fsig(ig[r]) * ftanh(gg[r]);
                cst[j][r]  = cv;
                hreg[j][r] = fsig(og[r]) * ftanh(cv);
            }
        }
        // h_t -> staging (same-wave DS ops are in-order: this step's reads already done)
#pragma unroll
        for (int j = 0; j < NH; ++j)
#pragma unroll
            for (int r = 0; r < 8; ++r) {
                const _Float16 hv = (_Float16)hreg[j][r];
                hst[(r + m8) * HS + j * 16 + nloc] = hv;
                if (WRITE_SEQ)
                    seq_out[(hrow[r] + t) * H + j * 16 + nloc] = hv;
            }
    }
    if (!WRITE_SEQ) {
#pragma unroll
        for (int j = 0; j < NH; ++j)
#pragma unroll
            for (int r = 0; r < 8; ++r)
                hfinal[(size_t)(b0 + r + m8) * H + j * 16 + nloc] = hreg[j][r];
    }
}

// ===== Head: hn[B,16] -> fc1(128) -> fc2(32) -> cls(6) =====
template <int WAVES>
__global__ __launch_bounds__(32 * WAVES)
void head_kernel(const float* __restrict__ hn,
                 const float* __restrict__ W1, const float* __restrict__ b1,
                 const float* __restrict__ W2, const float* __restrict__ b2,
                 const float* __restrict__ W3, const float* __restrict__ b3,
                 float* __restrict__ out) {
    __shared__ _Float16 s1_all[WAVES][16 * 128];
    __shared__ _Float16 s2_all[WAVES][16 * 32];

    const int lane = threadIdx.x & 31;
    const int wave = threadIdx.x >> 5;
    const int b0   = (blockIdx.x * WAVES + wave) * 16;
    const int nloc = lane & 15;
    const int m8   = (lane >> 4) << 3;
    _Float16* s1 = s1_all[wave];
    _Float16* s2 = s2_all[wave];

    v16h A;
    {
        const int hi = (lane >= 16) ? 8 : 0;
#pragma unroll
        for (int e = 0; e < 16; ++e) {
            const int r = e >> 1;
            const int k = ((r & 4) ? 16 : 0) + hi + (r & 3) * 2 + (e & 1);
            A[e] = (_Float16)((k < 16) ? hn[(size_t)(b0 + nloc) * 16 + k] : 0.0f);
        }
    }
#pragma unroll
    for (int nt = 0; nt < 8; ++nt) {
        v16h Bf;
#pragma unroll
        for (int e = 0; e < 16; ++e)
            Bf[e] = (_Float16)((e < 8) ? W1[(nt * 16 + nloc) * 16 + m8 + e] : 0.0f);
        v8f a = splat8(b1[nt * 16 + nloc]);
        a = wmma16(A, Bf, a);
#pragma unroll
        for (int r = 0; r < 8; ++r)
            s1[(r + m8) * 128 + nt * 16 + nloc] = (_Float16)a[r];
    }
    v8f a2[2];
#pragma unroll
    for (int nt = 0; nt < 2; ++nt) a2[nt] = splat8(b2[nt * 16 + nloc]);
#pragma unroll
    for (int s = 0; s < 4; ++s) {
        const v16h As = load_a_frag<128>(s1, nloc, s * 32, lane);
#pragma unroll
        for (int nt = 0; nt < 2; ++nt) {
            v16h Bf;
#pragma unroll
            for (int e = 0; e < 16; ++e)
                Bf[e] = (_Float16)W2[(nt * 16 + nloc) * 128 + s * 32 + ((e & 8) ? 16 : 0) + m8 + (e & 7)];
            a2[nt] = wmma16(As, Bf, a2[nt]);
        }
    }
#pragma unroll
    for (int nt = 0; nt < 2; ++nt)
#pragma unroll
        for (int r = 0; r < 8; ++r)
            s2[(r + m8) * 32 + nt * 16 + nloc] = (_Float16)a2[nt][r];

    const v16h A3 = load_a_frag<32>(s2, nloc, 0, lane);
    v16h B3;
#pragma unroll
    for (int e = 0; e < 16; ++e)
        B3[e] = (_Float16)((nloc < 6) ? W3[nloc * 32 + ((e & 8) ? 16 : 0) + m8 + (e & 7)] : 0.0f);
    v8f a3 = splat8((nloc < 6) ? b3[nloc] : 0.0f);
    a3 = wmma16(A3, B3, a3);

    if (nloc < 6) {
#pragma unroll
        for (int r = 0; r < 8; ++r)
            out[(size_t)(b0 + r + m8) * 6 + nloc] = a3[r];
    }
}

extern "C" void kernel_launch(void* const* d_in, const int* in_sizes, int n_in,
                              void* d_out, int out_size, void* d_ws, size_t ws_size,
                              hipStream_t stream) {
    const float* x     = (const float*)d_in[0];
    const float* Wih1  = (const float*)d_in[1];
    const float* Whh1  = (const float*)d_in[2];
    const float* bih1  = (const float*)d_in[3];
    const float* bhh1  = (const float*)d_in[4];
    const float* Wih2  = (const float*)d_in[5];
    const float* Whh2  = (const float*)d_in[6];
    const float* bih2  = (const float*)d_in[7];
    const float* bhh2  = (const float*)d_in[8];
    const float* Wih3  = (const float*)d_in[9];
    const float* Whh3  = (const float*)d_in[10];
    const float* bih3  = (const float*)d_in[11];
    const float* bhh3  = (const float*)d_in[12];
    const float* W_fc1 = (const float*)d_in[13];
    const float* b_fc1 = (const float*)d_in[14];
    const float* W_fc2 = (const float*)d_in[15];
    const float* b_fc2 = (const float*)d_in[16];
    const float* W_cls = (const float*)d_in[17];
    const float* b_cls = (const float*)d_in[18];

    const int T = 200, F = 36;
    const int B = in_sizes[0] / (T * F);   // 4096

    // Workspace: seq1 [B,T,16] f16 | seq2 [B,T,32] f16 | hfinal [B,16] f32
    char* ws = (char*)d_ws;
    _Float16* seq1 = (_Float16*)ws;
    _Float16* seq2 = (_Float16*)(ws + (size_t)B * T * 16 * sizeof(_Float16));
    float*    hfin = (float*)(ws + (size_t)B * T * 16 * sizeof(_Float16)
                                 + (size_t)B * T * 32 * sizeof(_Float16));

    constexpr int WAVES = 4;
    const int blocks = B / (16 * WAVES);
    const dim3 blk(32 * WAVES);

    lstm_layer_f32_kernel<16, 36, WAVES><<<blocks, blk, 0, stream>>>(
        x, Wih1, Whh1, bih1, bhh1, seq1, T);
    lstm_layer_tdm_kernel<32, 16, 10, true, WAVES><<<blocks, blk, 0, stream>>>(
        seq1, Wih2, Whh2, bih2, bhh2, seq2, nullptr, T);
    lstm_layer_tdm_kernel<16, 32, 10, false, WAVES><<<blocks, blk, 0, stream>>>(
        seq2, Wih3, Whh3, bih3, bhh3, nullptr, hfin, T);
    head_kernel<WAVES><<<blocks, blk, 0, stream>>>(
        hfin, W_fc1, b_fc1, W_fc2, b_fc2, W_cls, b_cls, (float*)d_out);
}
